// VariationalGCNEncoder_7765300871249
// MI455X (gfx1250) — matrix operand order repaired and checked by reading, not compiled, not run on hardware
//
#include <hip/hip_runtime.h>

// ---------------------------------------------------------------------------
// VariationalGCNEncoder for MI455X (gfx1250, wave32, WMMA).
// GEMMs: bf16 WMMA (v_wmma_f32_16x16x32_bf16). A staged via LDS with
// UNCONDITIONAL clamped loads, pipelined: 4 clause-able b128 loads first,
// then packed cvt + b64 DS stores. B pre-packed transposed bf16 so each
// fragment is two 16B per-lane global loads, 32B apart (clause-paired b128,
// cache resident). Graph aggregations use f32 global atomics (working set
// is L2-resident). Capture-safe.
// ---------------------------------------------------------------------------

typedef __attribute__((ext_vector_type(16))) __bf16 v16bf;
typedef __attribute__((ext_vector_type(8)))  __bf16 v8bf;
typedef __attribute__((ext_vector_type(4)))  __bf16 v4bf;
typedef __attribute__((ext_vector_type(8)))  float  v8f;

#define IN_C   256
#define HID    256
#define HEADS  8
#define HEAD_C 32
#define OUT_C  64
#define NEG_SLOPE 0.2f

// ------------------------------ helpers ------------------------------------

__device__ __forceinline__ float leaky(float x) { return x > 0.f ? x : NEG_SLOPE * x; }

// monotonic float<->uint mapping so atomicMax(u32) implements float max
__device__ __forceinline__ unsigned encf(float f) {
  unsigned u = __float_as_uint(f);
  return (u & 0x80000000u) ? ~u : (u | 0x80000000u);
}
__device__ __forceinline__ float decf(unsigned e) {
  return __uint_as_float((e & 0x80000000u) ? (e & 0x7FFFFFFFu) : ~e);
}

// ------------------------- weight pre-pack ----------------------------------
// w [K][Nc] f32 row-major  ->  wT [Nc][K] bf16 row-major
__global__ void k_pack_wT(const float* __restrict__ w, __bf16* __restrict__ wT,
                          int K, int Nc) {
  int idx = blockIdx.x * 256 + threadIdx.x;
  if (idx >= K * Nc) return;
  int k = idx / Nc, n = idx - k * Nc;           // coalesced read along n
  wT[(size_t)n * K + k] = (__bf16)w[idx];
}

// ------------------------------ WMMA GEMM ----------------------------------
// C[M,Nc] = A[M,K] * B[K,Nc]; A f32 row-major, B given as BT bf16 [Nc][K].
// Block: 256 threads = 8 waves. Tile BM=128, BN=64, BK=32.
// Wave w computes rows [w*16, w*16+16) x 64 cols = 4 WMMA accumulators.
// Out-of-range A rows are CLAMPED (not guarded): they only feed C rows >= M,
// which are never stored, so staging is branch-free.
__global__ __launch_bounds__(256)
void gemm_bf16_wmma(const float* __restrict__ A, const __bf16* __restrict__ BT,
                    float* __restrict__ C, int M, int K, int Nc) {
  __shared__ __bf16 As[128][32];   // [m][k]

  const int tid  = threadIdx.x;
  const int wave = tid >> 5;
  const int lane = tid & 31;
  const int lm   = lane & 15;      // 0..15
  const int lg   = lane >> 4;      // 0 or 1
  const int row0 = blockIdx.x * 128;
  const int col0 = blockIdx.y * 64;

  // clamped source rows for the 4 staging vectors owned by this thread
  int srow[4];
#pragma unroll
  for (int i = 0; i < 4; ++i) {
    int idx = tid + i * 256;               // 0..1023
    int r   = idx >> 3;                    // 0..127
    int gr  = row0 + r;
    srow[i] = gr < M ? gr : (M - 1);
  }
  const int skcol = (tid & 7) << 2;        // 0,4,...,28 (k offset within tile)

  v8f acc[4] = {};

  for (int k0 = 0; k0 < K; k0 += 32) {
    // phase 1: issue all 4 staging loads (clause-able, overlapped latency)
    float4 vbuf[4];
#pragma unroll
    for (int i = 0; i < 4; ++i)
      vbuf[i] = *(const float4*)&A[(size_t)srow[i] * K + (k0 + skcol)];

    __syncthreads();               // As WAR guard vs previous iteration

    // phase 2: packed cvt + b64 DS stores
#pragma unroll
    for (int i = 0; i < 4; ++i) {
      int r = (tid + i * 256) >> 3;
      v4bf o;
      o[0] = (__bf16)vbuf[i].x; o[1] = (__bf16)vbuf[i].y;
      o[2] = (__bf16)vbuf[i].z; o[3] = (__bf16)vbuf[i].w;
      *(v4bf*)&As[r][skcol] = o;
    }
    if (k0 + 32 < K)
      __builtin_prefetch(&A[(size_t)srow[0] * K + (k0 + 32) + skcol], 0, 1);
    __syncthreads();

    // A fragment (ISA 16-bit A 16x32 layout): two contiguous 16B LDS reads.
    v16bf af;
    {
      int m = wave * 16 + lm;
      v8bf lo = *(const v8bf*)&As[m][lg * 8];        // K = lg*8 .. +7
      v8bf hi = *(const v8bf*)&As[m][16 + lg * 8];   // K = 16+lg*8 .. +7
#pragma unroll
      for (int i = 0; i < 8; ++i) { af[i] = lo[i]; af[8 + i] = hi[i]; }
    }

#pragma unroll
    for (int j = 0; j < 4; ++j) {
      // B fragment (ISA 16-bit B 32x16 layout): two 16B per-lane loads
      // (K = lg*8..+7 and 16+lg*8..+7) from the pre-packed transposed weights.
      v16bf bfj;
      {
        const __bf16* bp = &BT[(size_t)(col0 + j * 16 + lm) * K + k0 + lg * 8];
        v8bf blo = *(const v8bf*)bp;
        v8bf bhi = *(const v8bf*)(bp + 16);
#pragma unroll
        for (int i = 0; i < 8; ++i) { bfj[i] = blo[i]; bfj[8 + i] = bhi[i]; }
      }
      acc[j] = __builtin_amdgcn_wmma_f32_16x16x32_bf16(
          /*neg_a=*/false, af, /*neg_b=*/false, bfj,
          /*c_mod=*/(short)0, acc[j], /*reuse_a=*/false, /*reuse_b=*/false);
    }
  }

  // store: C/D layout — VGPR r holds (M = r + 8*(lane/16), N = lane%16)
  if (row0 + 128 <= M) {
    // fast path: whole tile in range, no per-element guard
#pragma unroll
    for (int j = 0; j < 4; ++j) {
      int n = col0 + j * 16 + lm;
#pragma unroll
      for (int r = 0; r < 8; ++r) {
        int m = row0 + wave * 16 + r + (lg << 3);
        C[(size_t)m * Nc + n] = acc[j][r];
      }
    }
  } else {
#pragma unroll
    for (int j = 0; j < 4; ++j) {
      int n = col0 + j * 16 + lm;
#pragma unroll
      for (int r = 0; r < 8; ++r) {
        int m = row0 + wave * 16 + r + (lg << 3);
        if (m < M) C[(size_t)m * Nc + n] = acc[j][r];
      }
    }
  }
}

// --------------------------- graph kernels ---------------------------------

__global__ void k_fill_u32(unsigned* p, int n, unsigned v) {
  int i = blockIdx.x * 256 + threadIdx.x;
  if (i < n) p[i] = v;
}

__global__ void k_deg(const int* __restrict__ dst, float* __restrict__ deg, int E) {
  int i = blockIdx.x * 256 + threadIdx.x;
  if (i < E) atomicAdd(&deg[dst[i]], 1.0f);
}

__global__ void k_dinv(float* deg, int n) {
  int i = blockIdx.x * 256 + threadIdx.x;
  if (i < n) deg[i] = rsqrtf(deg[i] + 1.0f);   // +1 self loop
}

// agg[dst] += h[src]*dinv[src]*dinv[dst]; 4 edges per block, float4 per thread
__global__ __launch_bounds__(256)
void k_gcn_edge(const int* __restrict__ src, const int* __restrict__ dst,
                const float* __restrict__ dinv, const float* __restrict__ h,
                float* __restrict__ agg, int E) {
  int e = blockIdx.x * 4 + (threadIdx.x >> 6);
  if (e >= E) return;
  int l64 = threadIdx.x & 63;
  int s = src[e], d = dst[e];
  float norm = dinv[s] * dinv[d];
  int c4 = l64 << 2;
  float4 hv = *(const float4*)&h[(size_t)s * HID + c4];
  float* p = &agg[(size_t)d * HID + c4];
  atomicAdd(p + 0, hv.x * norm);
  atomicAdd(p + 1, hv.y * norm);
  atomicAdd(p + 2, hv.z * norm);
  atomicAdd(p + 3, hv.w * norm);
}

// h1 = gelu(agg + h*dinv^2 + b)  in place into agg
__global__ __launch_bounds__(256)
void k_finalize_gelu(float* __restrict__ agg, const float* __restrict__ h,
                     const float* __restrict__ dinv, const float* __restrict__ b, int N) {
  int i = blockIdx.x, c = threadIdx.x;
  float di = dinv[i];
  size_t o = (size_t)i * HID + c;
  float v = agg[o] + h[o] * di * di + b[c];
  float t = tanhf(0.7978845608028654f * (v + 0.044715f * v * v * v));
  agg[o] = 0.5f * v * (1.0f + t);
}

// per-node head scores (wave32: one wave == one head, lane == channel)
__global__ __launch_bounds__(256)
void k_scores(const float* __restrict__ hh, const float* __restrict__ att_s,
              const float* __restrict__ att_d, float* __restrict__ a_s,
              float* __restrict__ a_d, int N) {
  int i = blockIdx.x, t = threadIdx.x;
  float x = hh[(size_t)i * HID + t];
  float ps = x * att_s[t];
  float pd = x * att_d[t];
  for (int o = 16; o > 0; o >>= 1) {
    ps += __shfl_down(ps, o, 32);
    pd += __shfl_down(pd, o, 32);
  }
  if ((t & 31) == 0) {
    int h = t >> 5;
    a_s[i * HEADS + h] = ps;
    a_d[i * HEADS + h] = pd;
  }
}

// segment max over dst (edges + self loops): one thread per edge, 8 heads
__global__ void k_att_max(const int* __restrict__ src, const int* __restrict__ dst,
                          const float* __restrict__ a_s, const float* __restrict__ a_d,
                          unsigned* __restrict__ m_enc, int E, int N) {
  int e = blockIdx.x * 256 + threadIdx.x;
  if (e >= E + N) return;
  int s, d;
  if (e < E) { s = src[e]; d = dst[e]; } else { s = d = e - E; }
  const float4* as4 = (const float4*)&a_s[(size_t)s * HEADS];
  const float4* ad4 = (const float4*)&a_d[(size_t)d * HEADS];
  float4 s0 = as4[0], s1 = as4[1], d0 = ad4[0], d1 = ad4[1];
  float ev[HEADS] = { leaky(s0.x + d0.x), leaky(s0.y + d0.y),
                      leaky(s0.z + d0.z), leaky(s0.w + d0.w),
                      leaky(s1.x + d1.x), leaky(s1.y + d1.y),
                      leaky(s1.z + d1.z), leaky(s1.w + d1.w) };
  unsigned* mp = &m_enc[(size_t)d * HEADS];
#pragma unroll
  for (int h = 0; h < HEADS; ++h) atomicMax(mp + h, encf(ev[h]));
}

__global__ void k_decode_m(const unsigned* __restrict__ in, float* __restrict__ out, int n) {
  int i = blockIdx.x * 256 + threadIdx.x;
  if (i < n) { float v = decf(in[i]); out[i] = v; }
}

// ssum[dst,h] += exp(e - m[dst,h]): one thread per edge, 8 heads
__global__ void k_att_sum(const int* __restrict__ src, const int* __restrict__ dst,
                          const float* __restrict__ a_s, const float* __restrict__ a_d,
                          const float* __restrict__ m, float* __restrict__ ssum,
                          int E, int N) {
  int e = blockIdx.x * 256 + threadIdx.x;
  if (e >= E + N) return;
  int s, d;
  if (e < E) { s = src[e]; d = dst[e]; } else { s = d = e - E; }
  const float4* as4 = (const float4*)&a_s[(size_t)s * HEADS];
  const float4* ad4 = (const float4*)&a_d[(size_t)d * HEADS];
  const float4* mm4 = (const float4*)&m[(size_t)d * HEADS];
  float4 s0 = as4[0], s1 = as4[1], d0 = ad4[0], d1 = ad4[1];
  float4 m0 = mm4[0], m1 = mm4[1];
  float ex[HEADS] = { expf(leaky(s0.x + d0.x) - m0.x), expf(leaky(s0.y + d0.y) - m0.y),
                      expf(leaky(s0.z + d0.z) - m0.z), expf(leaky(s0.w + d0.w) - m0.w),
                      expf(leaky(s1.x + d1.x) - m1.x), expf(leaky(s1.y + d1.y) - m1.y),
                      expf(leaky(s1.z + d1.z) - m1.z), expf(leaky(s1.w + d1.w) - m1.w) };
  float* sp = &ssum[(size_t)d * HEADS];
#pragma unroll
  for (int h = 0; h < HEADS; ++h) atomicAdd(sp + h, ex[h]);
}

// agg[dst] += hh[src]*alpha; 4 edges per block, float4 per thread
__global__ __launch_bounds__(256)
void k_att_agg(const int* __restrict__ src, const int* __restrict__ dst,
               const float* __restrict__ a_s, const float* __restrict__ a_d,
               const float* __restrict__ m, const float* __restrict__ ssum,
               const float* __restrict__ hh, float* __restrict__ agg, int E) {
  int e = blockIdx.x * 4 + (threadIdx.x >> 6);
  if (e >= E) return;
  int l64 = threadIdx.x & 63;
  int h = l64 >> 3;                       // 4 consecutive channels share a head
  int s = src[e], d = dst[e];
  float ev = leaky(a_s[s * HEADS + h] + a_d[d * HEADS + h]);
  float alpha = expf(ev - m[d * HEADS + h]) / ssum[d * HEADS + h];
  int c4 = l64 << 2;
  float4 hv = *(const float4*)&hh[(size_t)s * HID + c4];
  float* p = &agg[(size_t)d * HID + c4];
  atomicAdd(p + 0, hv.x * alpha);
  atomicAdd(p + 1, hv.y * alpha);
  atomicAdd(p + 2, hv.z * alpha);
  atomicAdd(p + 3, hv.w * alpha);
}

// h2 = relu(agg + hh*alpha_self + b_att)  in place into agg
__global__ __launch_bounds__(256)
void k_finalize_att(float* __restrict__ agg, const float* __restrict__ hh,
                    const float* __restrict__ a_s, const float* __restrict__ a_d,
                    const float* __restrict__ m, const float* __restrict__ ssum,
                    const float* __restrict__ b, int N) {
  int i = blockIdx.x, t = threadIdx.x, h = t >> 5;
  float ev = leaky(a_s[i * HEADS + h] + a_d[i * HEADS + h]);
  float alpha = expf(ev - m[i * HEADS + h]) / ssum[i * HEADS + h];
  size_t o = (size_t)i * HID + t;
  float v = agg[o] + hh[o] * alpha + b[t];
  agg[o] = v > 0.f ? v : 0.f;
}

// dual 64-wide GCN aggregation for mu / logstd: 8 edges per block
__global__ __launch_bounds__(256)
void k_gcn_edge2(const int* __restrict__ src, const int* __restrict__ dst,
                 const float* __restrict__ dinv, const float* __restrict__ gmu,
                 const float* __restrict__ gls, float* __restrict__ amu,
                 float* __restrict__ als, int E) {
  int e = blockIdx.x * 8 + (threadIdx.x >> 5);
  if (e >= E) return;
  int l32 = threadIdx.x & 31;
  int s = src[e], d = dst[e];
  float norm = dinv[s] * dinv[d];
  const float* g = (l32 < 16) ? gmu : gls;
  float* a       = (l32 < 16) ? amu : als;
  int c4 = (l32 & 15) << 2;
  float4 v = *(const float4*)&g[(size_t)s * OUT_C + c4];
  float* p = &a[(size_t)d * OUT_C + c4];
  atomicAdd(p + 0, v.x * norm);
  atomicAdd(p + 1, v.y * norm);
  atomicAdd(p + 2, v.z * norm);
  atomicAdd(p + 3, v.w * norm);
}

__global__ __launch_bounds__(128)
void k_finalize_out(const float* __restrict__ amu, const float* __restrict__ als,
                    const float* __restrict__ gmu, const float* __restrict__ gls,
                    const float* __restrict__ dinv, const float* __restrict__ bmu,
                    const float* __restrict__ bls, float* __restrict__ out, int N) {
  int i = blockIdx.x, t = threadIdx.x;
  float di = dinv[i];
  float sl = di * di;
  if (t < OUT_C) {
    out[(size_t)i * OUT_C + t] =
        amu[(size_t)i * OUT_C + t] + gmu[(size_t)i * OUT_C + t] * sl + bmu[t];
  } else {
    int c = t - OUT_C;
    out[(size_t)N * OUT_C + (size_t)i * OUT_C + c] =
        als[(size_t)i * OUT_C + c] + gls[(size_t)i * OUT_C + c] * sl + bls[c];
  }
}

// ------------------------------ launcher -----------------------------------

extern "C" void kernel_launch(void* const* d_in, const int* in_sizes, int n_in,
                              void* d_out, int out_size, void* d_ws, size_t ws_size,
                              hipStream_t stream) {
  const float* x     = (const float*)d_in[0];
  const int*   ei    = (const int*)d_in[1];
  const float* w1    = (const float*)d_in[2];
  const float* b1    = (const float*)d_in[3];
  const float* w_att = (const float*)d_in[4];
  const float* att_s = (const float*)d_in[5];
  const float* att_d = (const float*)d_in[6];
  const float* b_att = (const float*)d_in[7];
  const float* w_mu  = (const float*)d_in[8];
  const float* b_mu  = (const float*)d_in[9];
  const float* w_ls  = (const float*)d_in[10];
  const float* b_ls  = (const float*)d_in[11];
  float* out = (float*)d_out;

  const int N = in_sizes[0] / IN_C;
  const int E = in_sizes[1] / 2;
  const int* src = ei;        // edge_index[0]
  const int* dst = ei + E;    // edge_index[1]

  // workspace layout (f32 elements)
  float* bufB = (float*)d_ws;                        // N*HID
  float* bufA = bufB + (size_t)N * HID;              // N*HID
  float* dinv = bufA + (size_t)N * HID;              // N
  float* a_s  = dinv + N;                            // N*HEADS
  float* a_d  = a_s + (size_t)N * HEADS;             // N*HEADS
  unsigned* m_enc = (unsigned*)(a_d + (size_t)N * HEADS);  // N*HEADS
  float* m_f  = (float*)m_enc;                       // decoded in place
  float* ssum = (float*)(m_enc + (size_t)N * HEADS); // N*HEADS
  __bf16* w1T  = (__bf16*)(ssum + (size_t)N * HEADS);      // IN_C*HID bf16
  __bf16* waT  = w1T + (size_t)IN_C * HID;                 // HID*HID bf16
  __bf16* wmT  = waT + (size_t)HID * HID;                  // OUT_C*HID bf16
  __bf16* wlT  = wmT + (size_t)HID * OUT_C;                // OUT_C*HID bf16

  // stage-3 sub-buffers inside bufB (4 * N*OUT_C == N*HID)
  float* gmu = bufB;
  float* gls = bufB + (size_t)N * OUT_C;
  float* amu = bufB + (size_t)2 * N * OUT_C;
  float* als = bufB + (size_t)3 * N * OUT_C;

  // ---- pack all weights to bf16 transposed --------------------------------
  k_pack_wT<<<(IN_C * HID + 255) / 256, 256, 0, stream>>>(w1, w1T, IN_C, HID);
  k_pack_wT<<<(HID * HID + 255) / 256, 256, 0, stream>>>(w_att, waT, HID, HID);
  k_pack_wT<<<(HID * OUT_C + 255) / 256, 256, 0, stream>>>(w_mu, wmT, HID, OUT_C);
  k_pack_wT<<<(HID * OUT_C + 255) / 256, 256, 0, stream>>>(w_ls, wlT, HID, OUT_C);

  // ---- degrees & symmetric norm -------------------------------------------
  hipMemsetAsync(dinv, 0, (size_t)N * sizeof(float), stream);
  k_deg<<<(E + 255) / 256, 256, 0, stream>>>(dst, dinv, E);
  k_dinv<<<(N + 255) / 256, 256, 0, stream>>>(dinv, N);

  // ---- layer 1: GCN(256->256) + gelu --------------------------------------
  dim3 gHID((N + 127) / 128, HID / 64);
  gemm_bf16_wmma<<<gHID, 256, 0, stream>>>(x, w1T, bufB, N, IN_C, HID);   // h
  hipMemsetAsync(bufA, 0, (size_t)N * HID * sizeof(float), stream);
  k_gcn_edge<<<(E + 3) / 4, 256, 0, stream>>>(src, dst, dinv, bufB, bufA, E);
  k_finalize_gelu<<<N, 256, 0, stream>>>(bufA, bufB, dinv, b1, N);        // h1

  // ---- layer 2: 8-head GAT + relu -----------------------------------------
  gemm_bf16_wmma<<<gHID, 256, 0, stream>>>(bufA, waT, bufB, N, HID, HID); // hh
  k_scores<<<N, 256, 0, stream>>>(bufB, att_s, att_d, a_s, a_d, N);

  const int totE = E + N;
  k_fill_u32<<<(N * HEADS + 255) / 256, 256, 0, stream>>>(m_enc, N * HEADS,
                                                          0x007FFFFFu /* enc(-inf) */);
  k_att_max<<<(totE + 255) / 256, 256, 0, stream>>>(src, dst, a_s, a_d, m_enc, E, N);
  k_decode_m<<<(N * HEADS + 255) / 256, 256, 0, stream>>>(m_enc, m_f, N * HEADS);
  hipMemsetAsync(ssum, 0, (size_t)N * HEADS * sizeof(float), stream);
  k_att_sum<<<(totE + 255) / 256, 256, 0, stream>>>(src, dst, a_s, a_d, m_f, ssum, E, N);

  hipMemsetAsync(bufA, 0, (size_t)N * HID * sizeof(float), stream);
  k_att_agg<<<(E + 3) / 4, 256, 0, stream>>>(src, dst, a_s, a_d, m_f, ssum, bufB, bufA, E);
  k_finalize_att<<<N, 256, 0, stream>>>(bufA, bufB, a_s, a_d, m_f, ssum, b_att, N); // h2

  // ---- layer 3: two GCN heads (256->64) -----------------------------------
  dim3 gOUT((N + 127) / 128, OUT_C / 64);
  gemm_bf16_wmma<<<gOUT, 256, 0, stream>>>(bufA, wmT, gmu, N, HID, OUT_C);
  gemm_bf16_wmma<<<gOUT, 256, 0, stream>>>(bufA, wlT, gls, N, HID, OUT_C);

  hipMemsetAsync(amu, 0, (size_t)2 * N * OUT_C * sizeof(float), stream);
  k_gcn_edge2<<<(E + 7) / 8, 256, 0, stream>>>(src, dst, dinv, gmu, gls, amu, als, E);
  k_finalize_out<<<N, 128, 0, stream>>>(amu, als, gmu, gls, dinv, b_mu, b_ls, out, N);
}
